// GINBlock_49581102465153
// MI455X (gfx1250) — matrix operand order repaired and checked by reading, not compiled, not run on hardware
//
#include <hip/hip_runtime.h>
#include <hip/hip_bf16.h>

// ---------------------------------------------------------------------------
// GIN block: out = relu(bn3( relu(bn2( relu(bn1((x+agg)@W1))@W2 )) ))
// (biases b1/b2 dropped: a per-feature constant added before a training-mode
//  BatchNorm shifts the batch mean by exactly that constant -> identical output)
//
// Memory-bound (~620 MB traffic @ 23.3 TB/s ~ 27us); GEMMs are ~4 GFLOP, i.e.
// free, so they run through v_wmma_f32_16x16x32_bf16 with a hi/lo bf16 split
// (3 WMMAs per K-step) for near-f32 accuracy.
// BN stats use 64-way binned atomics to kill same-address RMW serialization.
// ---------------------------------------------------------------------------

typedef __bf16  v16bf __attribute__((ext_vector_type(16)));
typedef float   v8f   __attribute__((ext_vector_type(8)));

#define BN_EPS 1e-5f
#define DH 128
#define DIN 32
#define NBINS 64

__device__ __forceinline__ void split_bf16(const float* a, v16bf& hi, v16bf& lo) {
#pragma unroll
    for (int i = 0; i < 16; ++i) {
        __bf16 h = (__bf16)a[i];
        hi[i] = h;
        lo[i] = (__bf16)(a[i] - (float)h);
    }
}

__device__ __forceinline__ v8f wmma3(const v16bf& ahi, const v16bf& alo,
                                     const v16bf& bhi, const v16bf& blo, v8f acc) {
    acc = __builtin_amdgcn_wmma_f32_16x16x32_bf16(false, ahi, false, bhi, (short)0, acc, false, false);
    acc = __builtin_amdgcn_wmma_f32_16x16x32_bf16(false, ahi, false, blo, (short)0, acc, false, false);
    acc = __builtin_amdgcn_wmma_f32_16x16x32_bf16(false, alo, false, bhi, (short)0, acc, false, false);
    return acc;
}

// ---- zero the binned stats scratch (6 x NBINS x 128 floats) ----------------
__global__ void k_zero(float* stats, int n) {
    int i = blockIdx.x * blockDim.x + threadIdx.x;
    if (i < n) stats[i] = 0.f;
}

// ---- h0 = x (vectorized copy) ----------------------------------------------
__global__ void k_copy(const float4* __restrict__ x, float4* __restrict__ h0, int n4) {
    int i = blockIdx.x * blockDim.x + threadIdx.x;
    if (i < n4) h0[i] = x[i];
}

// ---- h0[dst] += x[src]  (one thread per (edge, feature)) -------------------
__global__ void k_scatter(const float* __restrict__ x, const int* __restrict__ src,
                          const int* __restrict__ dst, float* __restrict__ h0,
                          long long total) {
    long long t = (long long)blockIdx.x * blockDim.x + threadIdx.x;
    if (t >= total) return;
    int e = (int)(t >> 5);
    int f = (int)(t & 31);
    int s = src[e], d = dst[e];
    atomicAdd(&h0[(size_t)d * DIN + f], x[(size_t)s * DIN + f]);
}

// ---- GEMM1: h1pre = h0(Nx32) @ W1(32x128); binned per-feature stats --------
__global__ void k_gemm1(const float* __restrict__ h0, const float* __restrict__ W1,
                        float* __restrict__ h1pre, float* __restrict__ sumb,
                        float* __restrict__ ssqb, int N) {
    const int lane = threadIdx.x & 31;
    const int wv   = threadIdx.x >> 5;     // 8 waves -> 8 column tiles of 16
    const int half = lane >> 4;
    const int l    = lane & 15;
    const int r0   = blockIdx.x * 16;
    const int row  = r0 + l;               // A-matrix row for this lane
    const int kb   = 8 * half;
    const int c    = wv * 16 + l;          // global output column
    const bool full = (r0 + 16 <= N);      // uniform per block

    float a[16];
    if (full || row < N) {
        const float* pr = h0 + (size_t)row * DIN;
#pragma unroll
        for (int i = 0; i < 8; ++i) { a[i] = pr[kb + i]; a[8 + i] = pr[16 + kb + i]; }
    } else {
#pragma unroll
        for (int i = 0; i < 16; ++i) a[i] = 0.f;
    }
    float b[16];
#pragma unroll
    for (int i = 0; i < 16; ++i) b[i] = W1[(size_t)(kb + (i & 7) + 16 * (i >> 3)) * DH + c];

    v16bf ahi, alo, bhi, blo;
    split_bf16(a, ahi, alo);
    split_bf16(b, bhi, blo);
    v8f acc = {};
    acc = wmma3(ahi, alo, bhi, blo, acc);

    float s = 0.f, ss = 0.f;
    float* outp = h1pre + (size_t)(r0 + 8 * half) * DH + c;
    if (full) {                            // branch-free interior path
#pragma unroll
        for (int r = 0; r < 8; ++r) {
            float v = acc[r];
            outp[(size_t)r * DH] = v;
            s += v; ss += v * v;
        }
    } else {
#pragma unroll
        for (int r = 0; r < 8; ++r) {
            if (r0 + r + 8 * half < N) {
                float v = acc[r];
                outp[(size_t)r * DH] = v;
                s += v; ss += v * v;
            }
        }
    }
    s  += __shfl_xor(s, 16, 32);
    ss += __shfl_xor(ss, 16, 32);
    if (half == 0) {
        int bin = (blockIdx.x & (NBINS - 1)) * DH;
        atomicAdd(&sumb[bin + c], s);
        atomicAdd(&ssqb[bin + c], ss);
    }
}

// ---- finalize: reduce bins, (sum,sumsq) -> (scale,shift) -------------------
__global__ void k_finalize(const float* __restrict__ sumb, const float* __restrict__ ssqb,
                           const float* __restrict__ g, const float* __restrict__ be,
                           float* __restrict__ scale, float* __restrict__ shift,
                           float invN) {
    int f = threadIdx.x;
    if (f < DH) {
        float s = 0.f, ss = 0.f;
#pragma unroll 4
        for (int b = 0; b < NBINS; ++b) {
            s  += sumb[b * DH + f];
            ss += ssqb[b * DH + f];
        }
        float m  = s * invN;
        float v  = ss * invN - m * m;
        float sc = g[f] * rsqrtf(v + BN_EPS);
        scale[f] = sc;
        shift[f] = be[f] - m * sc;
    }
}

// ---- GEMM2: h2pre = relu(bn1(h1pre)) @ W2(128x128); binned stats -----------
__global__ void k_gemm2(const float* __restrict__ h1pre, const float* __restrict__ W2,
                        const float* __restrict__ scale1, const float* __restrict__ shift1,
                        float* __restrict__ h2pre, float* __restrict__ sumb,
                        float* __restrict__ ssqb, int N) {
    const int lane = threadIdx.x & 31;
    const int wv   = threadIdx.x >> 5;
    const int half = lane >> 4;
    const int l    = lane & 15;
    const int r0   = blockIdx.x * 16;
    const int row  = r0 + l;
    const int kb   = 8 * half;
    const int c    = wv * 16 + l;
    const bool full = (r0 + 16 <= N);      // uniform per block
    const bool rowok = full || row < N;
    const float* pr = h1pre + (size_t)row * DH;

    v8f acc = {};
#pragma unroll
    for (int ks = 0; ks < DH; ks += 32) {
        float a[16];
        if (rowok) {
#pragma unroll
            for (int i = 0; i < 8; ++i) {
                int k  = ks + kb + i;
                int k2 = k + 16;
                a[i]     = fmaxf(pr[k]  * scale1[k]  + shift1[k],  0.f);  // BN1+ReLU fused
                a[8 + i] = fmaxf(pr[k2] * scale1[k2] + shift1[k2], 0.f);
            }
        } else {
#pragma unroll
            for (int i = 0; i < 16; ++i) a[i] = 0.f;
        }
        float b[16];
#pragma unroll
        for (int i = 0; i < 16; ++i) b[i] = W2[(size_t)(ks + kb + (i & 7) + 16 * (i >> 3)) * DH + c];

        v16bf ahi, alo, bhi, blo;
        split_bf16(a, ahi, alo);
        split_bf16(b, bhi, blo);
        acc = wmma3(ahi, alo, bhi, blo, acc);
    }

    float s = 0.f, ss = 0.f;
    float* outp = h2pre + (size_t)(r0 + 8 * half) * DH + c;
    if (full) {
#pragma unroll
        for (int r = 0; r < 8; ++r) {
            float v = acc[r];
            outp[(size_t)r * DH] = v;
            s += v; ss += v * v;
        }
    } else {
#pragma unroll
        for (int r = 0; r < 8; ++r) {
            if (r0 + r + 8 * half < N) {
                float v = acc[r];
                outp[(size_t)r * DH] = v;
                s += v; ss += v * v;
            }
        }
    }
    s  += __shfl_xor(s, 16, 32);
    ss += __shfl_xor(ss, 16, 32);
    if (half == 0) {
        int bin = (blockIdx.x & (NBINS - 1)) * DH;
        atomicAdd(&sumb[bin + c], s);
        atomicAdd(&ssqb[bin + c], ss);
    }
}

// ---- h2 = relu(bn2(h2pre)); binned stats for bn3 ---------------------------
__global__ void k_bn2(const float* __restrict__ h2pre, const float* __restrict__ scale2,
                      const float* __restrict__ shift2, float* __restrict__ h2,
                      float* __restrict__ sumb, float* __restrict__ ssqb, int N) {
    __shared__ float ls[256], lss[256];
    const int t    = threadIdx.x;
    const int f    = t & 127;
    const int half = t >> 7;               // 0/1 -> rows 0-7 / 8-15 of the strip
    const int r0   = blockIdx.x * 16;
    const bool full = (r0 + 16 <= N);
    const float sc = scale2[f], sh = shift2[f];
    const float* inp  = h2pre + (size_t)(r0 + half * 8) * DH + f;
    float*       outp = h2    + (size_t)(r0 + half * 8) * DH + f;

    float s = 0.f, ss = 0.f;
    if (full) {
#pragma unroll
        for (int r = 0; r < 8; ++r) {
            float v = fmaxf(inp[(size_t)r * DH] * sc + sh, 0.f);
            outp[(size_t)r * DH] = v;
            s += v; ss += v * v;
        }
    } else {
#pragma unroll
        for (int r = 0; r < 8; ++r) {
            if (r0 + half * 8 + r < N) {
                float v = fmaxf(inp[(size_t)r * DH] * sc + sh, 0.f);
                outp[(size_t)r * DH] = v;
                s += v; ss += v * v;
            }
        }
    }
    ls[t] = s; lss[t] = ss;
    __syncthreads();
    if (t < 128) {
        int bin = (blockIdx.x & (NBINS - 1)) * DH;
        atomicAdd(&sumb[bin + f], ls[t] + ls[t + 128]);
        atomicAdd(&ssqb[bin + f], lss[t] + lss[t + 128]);
    }
}

// ---- out = relu(bn3(h2)) (vectorized) --------------------------------------
__global__ void k_bn3(const float4* __restrict__ h2, const float* __restrict__ scale3,
                      const float* __restrict__ shift3, float4* __restrict__ out, int n4) {
    int i = blockIdx.x * blockDim.x + threadIdx.x;
    if (i >= n4) return;
    int f0 = (i & 31) * 4;                 // 128 feats / 4 per vec = 32 vecs per row
    float4 v = h2[i];
    float4 o;
    o.x = fmaxf(v.x * scale3[f0 + 0] + shift3[f0 + 0], 0.f);
    o.y = fmaxf(v.y * scale3[f0 + 1] + shift3[f0 + 1], 0.f);
    o.z = fmaxf(v.z * scale3[f0 + 2] + shift3[f0 + 2], 0.f);
    o.w = fmaxf(v.w * scale3[f0 + 3] + shift3[f0 + 3], 0.f);
    out[i] = o;
}

extern "C" void kernel_launch(void* const* d_in, const int* in_sizes, int n_in,
                              void* d_out, int out_size, void* d_ws, size_t ws_size,
                              hipStream_t stream) {
    const float* x   = (const float*)d_in[0];
    const int*   src = (const int*)  d_in[1];
    const int*   dst = (const int*)  d_in[2];
    const float* W1  = (const float*)d_in[3];
    // d_in[4] = b1: dropped (no-op before training-mode BN)
    const float* g1  = (const float*)d_in[5];
    const float* be1 = (const float*)d_in[6];
    const float* W2  = (const float*)d_in[7];
    // d_in[8] = b2: dropped
    const float* g2  = (const float*)d_in[9];
    const float* be2 = (const float*)d_in[10];
    const float* g3  = (const float*)d_in[11];
    const float* be3 = (const float*)d_in[12];
    float* out = (float*)d_out;

    const int N = in_sizes[0] / DIN;
    const int E = in_sizes[1];
    const float invN = 1.0f / (float)N;

    // workspace layout (floats)
    float* ws    = (float*)d_ws;
    float* h0    = ws;                                  // N x 32
    float* h1    = h0 + (size_t)N * DIN;                // N x 128 (h1pre, reused as h2)
    float* h2pre = h1 + (size_t)N * DH;                 // N x 128
    float* stats = h2pre + (size_t)N * DH;              // binned + finalized stats
    const int BSZ = NBINS * DH;                         // 8192 floats per binned buf
    float* sum1b = stats;            float* ss1b = sum1b + BSZ;
    float* sum2b = ss1b + BSZ;       float* ss2b = sum2b + BSZ;
    float* sum3b = ss2b + BSZ;       float* ss3b = sum3b + BSZ;
    float* fin   = ss3b + BSZ;                          // 6 x 128
    float* sc1 = fin,        *sh1 = fin + 128;
    float* sc2 = fin + 256,  *sh2 = fin + 384;
    float* sc3 = fin + 512,  *sh3 = fin + 640;

    const int rb = (N + 15) / 16;                       // 16-row strips
    const int n4_h0  = N * DIN / 4;
    const int n4_h   = N * DH / 4;
    const int nzero  = 6 * BSZ;
    const long long sc_total = (long long)E * DIN;
    const int sc_blocks = (int)((sc_total + 255) / 256);

    k_zero   <<<(nzero + 255) / 256, 256, 0, stream>>>(stats, nzero);
    k_copy   <<<(n4_h0 + 255) / 256, 256, 0, stream>>>((const float4*)x, (float4*)h0, n4_h0);
    k_scatter<<<sc_blocks, 256, 0, stream>>>(x, src, dst, h0, sc_total);
    k_gemm1  <<<rb, 256, 0, stream>>>(h0, W1, h1, sum1b, ss1b, N);
    k_finalize<<<1, 128, 0, stream>>>(sum1b, ss1b, g1, be1, sc1, sh1, invN);
    k_gemm2  <<<rb, 256, 0, stream>>>(h1, W2, sc1, sh1, h2pre, sum2b, ss2b, N);
    k_finalize<<<1, 128, 0, stream>>>(sum2b, ss2b, g2, be2, sc2, sh2, invN);
    k_bn2    <<<rb, 256, 0, stream>>>(h2pre, sc2, sh2, h1 /*h2 reuses h1*/, sum3b, ss3b, N);
    k_finalize<<<1, 128, 0, stream>>>(sum3b, ss3b, g3, be3, sc3, sh3, invN);
    k_bn3    <<<(n4_h + 255) / 256, 256, 0, stream>>>((const float4*)h1, sc3, sh3, (float4*)out, n4_h);
}
